// MMDLoss_18124761989320
// MI455X (gfx1250) — compile-verified, matrix operand order checked by reading, not run on hardware
//
#include <hip/hip_runtime.h>
#include <hip/hip_fp16.h>
#include <math.h>

typedef _Float16 half8 __attribute__((ext_vector_type(8)));
typedef _Float16 v16h  __attribute__((ext_vector_type(16)));
typedef float    v8f   __attribute__((ext_vector_type(8)));

#define NROWS 8192
#define HALFN 4096
#define DDIM  64
#define NTILE 512                       // 8192 / 16
#define NTRI  (NTILE * (NTILE + 1) / 2) // 131328 upper-tri tiles (incl. diag)

// smallest inv-bandwidth (1/32) times log2(e): exp(-l2/32) = exp2(-l2 * KBASE)
#define KBASE (0.03125f * 1.44269504088896340f)

// ---------------------------------------------------------------------------
// Prep: concat(source,target) -> f16 copy in ws, row squared-norms computed
// FROM THE f16-ROUNDED values (so the WMMA Gram diagonal cancels exactly),
// and zero the scalar output.
// ---------------------------------------------------------------------------
__global__ __launch_bounds__(256) void mmd_prep_kernel(
    const float* __restrict__ src, const float* __restrict__ tgt,
    _Float16* __restrict__ th, float* __restrict__ sq, float* __restrict__ out)
{
    const int row = blockIdx.x * blockDim.x + threadIdx.x;
    if (row == 0) out[0] = 0.0f;
    if (row >= NROWS) return;
    const float* in = (row < HALFN) ? (src + row * DDIM)
                                    : (tgt + (row - HALFN) * DDIM);
    _Float16* orow = th + row * DDIM;
    float s = 0.0f;
#pragma unroll
    for (int d = 0; d < DDIM; ++d) {
        _Float16 h = (_Float16)in[d];
        orow[d] = h;
        float f = (float)h;
        s += f * f;
    }
    sq[row] = s;
}

// ---------------------------------------------------------------------------
// Main: one wave per 16x16 Gram tile (grid-strided over upper triangle).
// Two v_wmma_f32_16x16x32_f16 per tile (K = 64). The 12-bandwidth Gaussian
// sum uses ONE native v_exp_f32 plus 11 squarings (geometric bandwidths).
// ---------------------------------------------------------------------------
__global__ __launch_bounds__(256) void mmd_main_kernel(
    const _Float16* __restrict__ th, const float* __restrict__ sq,
    float* __restrict__ out)
{
    __shared__ float wsum[8];

    const int lane  = threadIdx.x & 31;
    const int wave  = threadIdx.x >> 5;
    const int wid   = blockIdx.x * (blockDim.x >> 5) + wave;
    const int wstep = gridDim.x * (blockDim.x >> 5);

    const int lm   = lane & 15;       // row-in-tile for A/B fragments, N for C
    const int lh8  = (lane >> 4) * 8; // K sub-offset for A/B, M offset for C

    float acc = 0.0f;

    for (int t = wid; t < NTRI; t += wstep) {
        // Decode upper-triangular tile index (ti <= tj), row-major.
        int r = NTRI - 1 - t;                      // index from the end
        int k = (int)((sqrtf((float)(8 * r + 1)) - 1.0f) * 0.5f);
        while ((k + 1) * (k + 2) / 2 <= r) ++k;    // integer fix-up
        while (k * (k + 1) / 2 > r) --k;
        const int ti = NTILE - 1 - k;
        const int tj = NTILE - 1 - (r - k * (k + 1) / 2);

        const int i0 = ti << 4, j0 = tj << 4;

        const _Float16* arow = th + (i0 + lm) * DDIM + lh8;
        const _Float16* brow = th + (j0 + lm) * DDIM + lh8;

        // Issue all 8 b128 loads up front, then both WMMAs.
        half8 a0lo = *(const half8*)(arow);
        half8 a0hi = *(const half8*)(arow + 16);
        half8 b0lo = *(const half8*)(brow);
        half8 b0hi = *(const half8*)(brow + 16);
        half8 a1lo = *(const half8*)(arow + 32);
        half8 a1hi = *(const half8*)(arow + 48);
        half8 b1lo = *(const half8*)(brow + 32);
        half8 b1hi = *(const half8*)(brow + 48);

        v16h a0, b0, a1, b1;
#pragma unroll
        for (int e = 0; e < 8; ++e) {
            a0[e] = a0lo[e]; a0[e + 8] = a0hi[e];
            b0[e] = b0lo[e]; b0[e + 8] = b0hi[e];
            a1[e] = a1lo[e]; a1[e + 8] = a1hi[e];
            b1[e] = b1lo[e]; b1[e + 8] = b1hi[e];
        }

        v8f c = {};
        c = __builtin_amdgcn_wmma_f32_16x16x32_f16(
                false, a0, false, b0, (short)0, c, false, false);
        c = __builtin_amdgcn_wmma_f32_16x16x32_f16(
                false, a1, false, b1, (short)0, c, false, false);

        // Epilogue: C layout -> lane holds (M = rr + lh8, N = lm)
        const float sqj = sq[j0 + lm];
        float tsum = 0.0f;
#pragma unroll
        for (int rr = 0; rr < 8; ++rr) {
            const float sqi = sq[i0 + rr + lh8];
            const float l2  = sqi + sqj - 2.0f * c[rr];
            // Largest-bandwidth term via one native exp2; the other 11
            // bandwidths follow by repeated squaring (inv_bw doubles).
            float e  = __builtin_amdgcn_exp2f(-l2 * KBASE);
            float ks = e;
#pragma unroll
            for (int q = 0; q < 11; ++q) { e = e * e; ks += e; }
            tsum += ks;
        }

        // Tile weight: sign(s_i * s_j), doubled off-diagonal (symmetry).
        float w = ((ti < NTILE / 2) == (tj < NTILE / 2)) ? 1.0f : -1.0f;
        if (ti != tj) w *= 2.0f;
        acc += w * tsum;
    }

    // wave32 reduction
#pragma unroll
    for (int m = 16; m > 0; m >>= 1)
        acc += __shfl_xor(acc, m, 32);
    if (lane == 0) wsum[wave] = acc;
    __syncthreads();
    if (wave == 0) {
        float v = (lane < 8) ? wsum[lane] : 0.0f;
#pragma unroll
        for (int m = 4; m > 0; m >>= 1)
            v += __shfl_xor(v, m, 32);
        if (lane == 0)
            atomicAdd(out, v * 0x1p-24f);          // / (4096^2)
    }
}

// ---------------------------------------------------------------------------
extern "C" void kernel_launch(void* const* d_in, const int* in_sizes, int n_in,
                              void* d_out, int out_size, void* d_ws, size_t ws_size,
                              hipStream_t stream) {
    const float* src = (const float*)d_in[0];
    const float* tgt = (const float*)d_in[1];
    float* out = (float*)d_out;

    _Float16* th = (_Float16*)d_ws;                       // 8192*64*2 = 1 MB
    float*    sq = (float*)((char*)d_ws + (size_t)NROWS * DDIM * sizeof(_Float16));

    mmd_prep_kernel<<<NROWS / 256, 256, 0, stream>>>(src, tgt, th, sq, out);
    mmd_main_kernel<<<2048, 256, 0, stream>>>(th, sq, out);
}
